// TurboQuantMSE_21801253994898
// MI455X (gfx1250) — matrix operand order repaired and checked by reading, not compiled, not run on hardware
//
#include <hip/hip_runtime.h>
#include <hip/hip_bf16.h>
#include <hip/hip_fp16.h>

typedef __attribute__((ext_vector_type(16))) _Float16 v16h;
typedef __attribute__((ext_vector_type(8)))  float    v8f;

#define TQ_BLOCK 128   // vector length / contraction size
#define TQ_WAVES 8     // waves per workgroup (256 threads, wave32)
#define TQ_ROWS  16    // vectors (rows) per wave = one WMMA M-tile
#define TQ_NT    8     // 8 N-tiles of 16 -> 128 output cols
#define TQ_KB    4     // 4 K-blocks of 32 -> K=128

// workspace layout (byte offsets)
#define WS_CB   0       // 256 x f32 Lloyd-Max codebook
#define WS_BD   1024    // 256 x f32 decision boundaries (255 used + sentinel)
#define WS_RF   2048    // 16384 x f16 : R   in wave32 WMMA B-fragment order
#define WS_RTF  34816   // 16384 x f16 : R^T in wave32 WMMA B-fragment order
#define WS_TBL  67584   // 1024 x u16 : quantizer cell -> boundary-count LUT

// searchsorted(BOUNDS, z), side='left': count of bounds < z, result in [0,255]
__device__ __forceinline__ int tq_bsearch(const float* __restrict__ bd, float z) {
  int lo = 0, hi = 255;
  #pragma unroll
  for (int i = 0; i < 8; ++i) {
    int mid = (lo + hi) >> 1;
    bool up = bd[mid] < z;
    lo = up ? (mid + 1) : lo;
    hi = up ? hi : mid;
  }
  return lo;
}

// ---------------------------------------------------------------------------
// Prep kernel: Lloyd-Max codebook (on-device, deterministic), folded rotation
// matrix R = diag(s0)*HN*diag(s1)*HN and its transpose (entries are exact
// multiples of 1/128 -> exact in f16), pre-swizzled in B-fragment order:
//   frag[((kb*8+nt)*32 + lane)*16 + h] = B[kb*32 + 8*(lane>>4) + (h&7) + 16*(h>>3)][nt*16 + (lane&15)]
// plus a 1024-cell LUT over [-4,4] for O(1) quantization (min boundary
// spacing ~0.017 > cell width 0.0078 -> <=1 boundary per cell).
// ---------------------------------------------------------------------------
__global__ __launch_bounds__(256) void tq_prep(const int* __restrict__ signs,
                                               float* __restrict__ cb,
                                               float* __restrict__ bd,
                                               _Float16* __restrict__ Rf,
                                               _Float16* __restrict__ Rtf,
                                               unsigned short* __restrict__ tbl)
{
  __shared__ float c[256];
  __shared__ float sbd[256];
  const int t = threadIdx.x;

  // init: Gaussian quantiles, ndtri((t+0.5)/256) = sqrt(2)*erfinv(2p-1)
  float p = (float)(2 * t + 1) * (1.0f / 512.0f);
  c[t] = 1.41421356237f * erfinvf(2.0f * p - 1.0f);
  __syncthreads();

  const float INV_SQRT2PI = 0.39894228040143267f;
  const float INV_SQRT2   = 0.70710678118f;
  for (int it = 0; it < 200; ++it) {
    float cm = (t > 0)   ? c[t - 1] : 0.0f;
    float c0 = c[t];
    float cp = (t < 255) ? c[t + 1] : 0.0f;
    float Blo = (t == 0)   ? -12.0f : 0.5f * (cm + c0);
    float Bhi = (t == 255) ?  12.0f : 0.5f * (c0 + cp);
    float num = (expf(-0.5f * Blo * Blo) - expf(-0.5f * Bhi * Bhi)) * INV_SQRT2PI;
    float den = fmaxf(0.5f * erfcf(-Bhi * INV_SQRT2) - 0.5f * erfcf(-Blo * INV_SQRT2), 1e-12f);
    __syncthreads();
    c[t] = num / den;
    __syncthreads();
  }
  cb[t] = c[t];
  float bdv = (t < 255) ? 0.5f * (c[t] + c[t + 1]) : 1e30f;
  bd[t]  = bdv;
  sbd[t] = bdv;
  __syncthreads();

  // quantizer LUT: tbl[g] = #boundaries < (-4 + g/128)
  for (int g = t; g < 1024; g += 256) {
    float edge = -4.0f + (float)g * (1.0f / 128.0f);
    tbl[g] = (unsigned short)tq_bsearch(sbd, edge);
  }

  // R[k][n] = s0_k * (1/128) * sum_j S(k,j)*s1_j*S(j,n); inner sum symmetric
  // in (k,n), so R^T[k][n] = R[n][k] = s0_n * same_sum / 128.
  for (int e = t; e < 16384; e += 256) {
    int h    = e & 15;
    int lane = (e >> 4) & 31;
    int nt   = (e >> 9) & 7;
    int kb   = e >> 12;
    int k = kb * 32 + 8 * (lane >> 4) + (h & 7) + 16 * (h >> 3);   // contraction row of B
    int n = nt * 16 + (lane & 15);            // output col of B
    int acc = 0;
    for (int j = 0; j < 128; ++j) {
      int s1  = 1 - 2 * signs[128 + j];
      int par = (__popc(k & j) + __popc(j & n)) & 1;   // S(k,j)*S(j,n) sign
      acc += (1 - 2 * par) * s1;
    }
    float sum = (float)acc * (1.0f / 128.0f);
    Rf[e]  = (_Float16)((float)(1 - 2 * signs[k]) * sum);   // R[k][n]
    Rtf[e] = (_Float16)((float)(1 - 2 * signs[n]) * sum);   // R^T[k][n]
  }
}

// O(1) searchsorted via cell LUT + two ordered probes (exact: <=1 boundary
// per cell, probes cover 2). bd must have sentinels at [255..257].
__device__ __forceinline__ int tq_quant(const float* __restrict__ bd,
                                        const unsigned short* __restrict__ tbl,
                                        float z) {
  int gi = (int)((z + 4.0f) * 128.0f);
  gi = gi < 0 ? 0 : (gi > 1023 ? 1023 : gi);
  int base = (int)tbl[gi];
  return base + (bd[base] < z ? 1 : 0) + (bd[base + 1] < z ? 1 : 0);
}

__device__ __forceinline__ v16h tq_ldfrag(const _Float16* __restrict__ p) {
  v16h r;
  #pragma unroll
  for (int h = 0; h < 16; ++h) r[h] = p[h];   // 32 contiguous bytes -> 2x b128
  return r;
}

// ---------------------------------------------------------------------------
// Main kernel: each wave processes one 16x128 tile (16 vectors).
//   Y = X @ R          (A straight from global w/ f32->f16 cvt, f16 WMMA, f32 acc)
//   per-row rms, Lloyd-Max quantize, least-squares gamma, re-quantize
//   OUT = Q @ R^T      (Q staged in LDS in A-fragment order)
// ---------------------------------------------------------------------------
__global__ __launch_bounds__(256) void tq_main(const float* __restrict__ x,
                                               float* __restrict__ out,
                                               const float* __restrict__ g_cb,
                                               const float* __restrict__ g_bd,
                                               const unsigned short* __restrict__ g_tbl,
                                               const _Float16* __restrict__ Rf,
                                               const _Float16* __restrict__ Rtf)
{
  // per-wave Q tile in wave32 WMMA A-fragment order: sQ[((wave*4+kb)*32+lane)*16 + h]
  __shared__ _Float16 sQ[TQ_WAVES * TQ_KB * 32 * 16];   // 8 KB
  __shared__ float s_cb[256];
  __shared__ float s_bd[260];                            // +sentinels
  __shared__ unsigned short s_tbl[1024];

  const int tid  = threadIdx.x;
  const int wave = tid >> 5;
  const int lane = tid & 31;
  const int lo   = lane & 15;
  const int hi   = lane >> 4;

  const long rowbase = (long)blockIdx.x * (TQ_WAVES * TQ_ROWS) + (long)wave * TQ_ROWS;
  const float* __restrict__ xin = x + rowbase * TQ_BLOCK;

  __builtin_prefetch(xin, 0, 0);          // gfx1250 global_prefetch of our tile

  s_cb[tid] = g_cb[tid];
  s_bd[tid] = g_bd[tid];
  if (tid < 4) s_bd[256 + tid] = 1e30f;
  for (int i = tid; i < 1024; i += 256) s_tbl[i] = g_tbl[i];

  // ---- matmul 1: Y = X @ R, A-fragments loaded directly from global ----
  // A layout per lane: row = lo; halfs 0..7 -> cols kb*32+8*hi .. +7,
  //                    halfs 8..15 -> cols kb*32+16+8*hi .. +7.
  v8f acc[TQ_NT] = {};
  #pragma unroll
  for (int kb = 0; kb < TQ_KB; ++kb) {
    const float* rp = xin + lo * TQ_BLOCK + kb * 32 + 8 * hi;
    const float4 c0 = ((const float4*)rp)[0];
    const float4 c1 = ((const float4*)rp)[1];
    const float4 c2 = ((const float4*)rp)[4];   // +16 floats
    const float4 c3 = ((const float4*)rp)[5];
    v16h a;
    a[0]  = (_Float16)c0.x; a[1]  = (_Float16)c0.y; a[2]  = (_Float16)c0.z; a[3]  = (_Float16)c0.w;
    a[4]  = (_Float16)c1.x; a[5]  = (_Float16)c1.y; a[6]  = (_Float16)c1.z; a[7]  = (_Float16)c1.w;
    a[8]  = (_Float16)c2.x; a[9]  = (_Float16)c2.y; a[10] = (_Float16)c2.z; a[11] = (_Float16)c2.w;
    a[12] = (_Float16)c3.x; a[13] = (_Float16)c3.y; a[14] = (_Float16)c3.z; a[15] = (_Float16)c3.w;
    #pragma unroll
    for (int nt = 0; nt < TQ_NT; ++nt) {
      const v16h b = tq_ldfrag(Rf + (((kb * 8 + nt) * 32) + lane) * 16);
      acc[nt] = __builtin_amdgcn_wmma_f32_16x16x32_f16(false, a, false, b,
                                                       (short)0, acc[nt],
                                                       false, false);
    }
  }
  __syncthreads();   // tables visible to all waves before first use

  // ---- per-row stats + two-stage Lloyd-Max quantization ----
  // C/D layout: VGPR r, lanes 0-15 -> row r, lanes 16-31 -> row r+8; col = nt*16+lo
  float gamma[8];
  #pragma unroll
  for (int r = 0; r < 8; ++r) {
    float ss = 0.0f;
    #pragma unroll
    for (int nt = 0; nt < TQ_NT; ++nt) { float y = acc[nt][r]; ss = fmaf(y, y, ss); }
    #pragma unroll
    for (int off = 1; off < 16; off <<= 1) ss += __shfl_xor(ss, off, 16);
    const float inv_rms = 1.0f / (sqrtf(ss * (1.0f / 128.0f)) + 1e-10f);

    float num = 0.0f, den = 0.0f;
    #pragma unroll
    for (int nt = 0; nt < TQ_NT; ++nt) {
      float y   = acc[nt][r];
      float rec = s_cb[tq_quant(s_bd, s_tbl, y * inv_rms)];
      num = fmaf(y, rec, num);
      den = fmaf(rec, rec, den);
    }
    #pragma unroll
    for (int off = 1; off < 16; off <<= 1) {
      num += __shfl_xor(num, off, 16);
      den += __shfl_xor(den, off, 16);
    }
    gamma[r] = num / (den + 1e-10f);
  }

  // ---- re-quantize with gamma, write Q into LDS in A-fragment layout ----
  _Float16* __restrict__ myQ = sQ + wave * (TQ_KB * 32 * 16);
  #pragma unroll
  for (int r = 0; r < 8; ++r) {
    const float g     = gamma[r];
    const float inv_g = 1.0f / (g + 1e-10f);
    const int   row   = r + 8 * hi;
    #pragma unroll
    for (int nt = 0; nt < TQ_NT; ++nt) {
      const float q  = s_cb[tq_quant(s_bd, s_tbl, acc[nt][r] * inv_g)] * g;
      const int   c  = nt * 16 + lo;
      const int   kb = c >> 5;
      const int   k  = c & 31;
      const int   hk = (k >> 3) & 1;
      const int   hb = (k & 7) + 8 * (k >> 4);
      myQ[((kb * 32) + (row + 16 * hk)) * 16 + hb] = (_Float16)q;
    }
  }
  // per-wave private LDS region + in-order DS within a wave: no barrier needed

  // ---- matmul 2: OUT = Q @ R^T ----
  #pragma unroll
  for (int nt = 0; nt < TQ_NT; ++nt)
    #pragma unroll
    for (int j = 0; j < 8; ++j) acc[nt][j] = 0.0f;
  #pragma unroll
  for (int kb = 0; kb < TQ_KB; ++kb) {
    const v16h a = tq_ldfrag(myQ + (kb * 32 + lane) * 16);
    #pragma unroll
    for (int nt = 0; nt < TQ_NT; ++nt) {
      const v16h b = tq_ldfrag(Rtf + (((kb * 8 + nt) * 32) + lane) * 16);
      acc[nt] = __builtin_amdgcn_wmma_f32_16x16x32_f16(false, a, false, b,
                                                       (short)0, acc[nt],
                                                       false, false);
    }
  }

  // ---- store (nt-inner keeps each row's 128 floats contiguous for WC) ----
  float* __restrict__ orow = out + rowbase * TQ_BLOCK;
  #pragma unroll
  for (int r = 0; r < 8; ++r) {
    const int row = r + 8 * hi;
    #pragma unroll
    for (int nt = 0; nt < TQ_NT; ++nt)
      orow[row * TQ_BLOCK + nt * 16 + lo] = acc[nt][r];
  }
}

extern "C" void kernel_launch(void* const* d_in, const int* in_sizes, int n_in,
                              void* d_out, int out_size, void* d_ws, size_t ws_size,
                              hipStream_t stream) {
  const float* x     = (const float*)d_in[0];
  const int*   signs = (const int*)d_in[1];
  float*       out   = (float*)d_out;
  char*        ws    = (char*)d_ws;

  float*          cb  = (float*)(ws + WS_CB);
  float*          bd  = (float*)(ws + WS_BD);
  _Float16*       Rf  = (_Float16*)(ws + WS_RF);
  _Float16*       Rtf = (_Float16*)(ws + WS_RTF);
  unsigned short* tbl = (unsigned short*)(ws + WS_TBL);

  // Deterministic: rebuilt every call (no caching), trivial cost.
  tq_prep<<<1, 256, 0, stream>>>(signs, cb, bd, Rf, Rtf, tbl);

  const long rows = (long)in_sizes[0] / TQ_BLOCK;        // 524288 for the given shape
  const int  grid = (int)(rows / (TQ_WAVES * TQ_ROWS));  // 4096 blocks x 256 threads
  tq_main<<<grid, 256, 0, stream>>>(x, out, cb, bd, tbl, Rf, Rtf);

  (void)n_in; (void)out_size; (void)ws_size;
}
